// GNN_55628416418157
// MI455X (gfx1250) — matrix operand order repaired and checked by reading, not compiled, not run on hardware
//
#include <hip/hip_runtime.h>
#include <stdint.h>

typedef __attribute__((ext_vector_type(16))) _Float16 v16h;
typedef __attribute__((ext_vector_type(8)))  float    v8f;

// ---------------------------------------------------------------------------
// Degree / normalization precompute (shared by both GCN layers)
// ---------------------------------------------------------------------------
__global__ void k_init_deg(float* __restrict__ deg, int n) {
  int i = blockIdx.x * blockDim.x + threadIdx.x;
  if (i < n) deg[i] = 1.0f;  // self-loop weight 1
}

__global__ void k_deg_scatter(const long long* __restrict__ dst,
                              const float* __restrict__ w,
                              float* __restrict__ deg, int nE) {
  int e = blockIdx.x * blockDim.x + threadIdx.x;
  if (e < nE) atomicAdd(&deg[dst[e]], w[e]);
}

__global__ void k_dinv(float* __restrict__ deg, int n) {
  int i = blockIdx.x * blockDim.x + threadIdx.x;
  if (i < n) {
    float d = deg[i];
    deg[i] = (d > 0.f) ? rsqrtf(d) : 0.f;  // in-place deg -> dinv
  }
}

__global__ void k_norm(const long long* __restrict__ src,
                       const long long* __restrict__ dst,
                       const float* __restrict__ w,
                       const float* __restrict__ dinv,
                       float* __restrict__ norm, int nE) {
  int e = blockIdx.x * blockDim.x + threadIdx.x;
  if (e < nE) norm[e] = dinv[src[e]] * w[e] * dinv[dst[e]];
}

// ---------------------------------------------------------------------------
// WMMA GEMM: Y[N x COUT] = X[N x 64] @ W[64 x COUT]   (f16 inputs, f32 acc)
// One wave per 16-row tile; wave holds all COUT/16 column accumulators to
// reuse the A fragment. Layouts per CDNA5 ISA 7.12.2 (wave32).
// ---------------------------------------------------------------------------
template <int COUT>
__global__ __launch_bounds__(128) void k_gemm(const float* __restrict__ X,
                                              const float* __restrict__ W,
                                              float* __restrict__ Y,
                                              int n, int nTiles) {
  const int wave = threadIdx.x >> 5;
  const int lane = threadIdx.x & 31;
  const int tile = blockIdx.x * 4 + wave;
  if (tile >= nTiles) return;             // wave-uniform: EXEC stays all-1s

  const int rlane = lane & 15;            // row (A) / col (B,C) within tile
  const int hi    = lane >> 4;            // half-wave selector
  const int row   = tile * 16 + rlane;
  const int rowc  = row < n ? row : n - 1;  // clamp loads (no EXEC divergence)
  const float* xrow = X + (size_t)rowc * 64;

  // A fragments for K-steps 0 (K=0..31) and 1 (K=32..63)
  // lanes 0-15 : elems 0-7 -> K=kb+0..7,  elems 8-15 -> K=kb+16..23
  // lanes 16-31: elems 0-7 -> K=kb+8..15, elems 8-15 -> K=kb+24..31
  v16h a[2];
  const int lo = hi * 8;
#pragma unroll
  for (int s = 0; s < 2; ++s) {
    const int kb = s * 32;
#pragma unroll
    for (int j = 0; j < 8; ++j) {
      a[s][j]     = (_Float16)xrow[kb + lo + j];
      a[s][8 + j] = (_Float16)xrow[kb + 16 + lo + j];
    }
  }

  constexpr int NT = COUT / 16;
  v8f acc[NT] = {};

#pragma unroll
  for (int s = 0; s < 2; ++s) {
#pragma unroll
    for (int t = 0; t < NT; ++t) {
      // B fragment: lane holds column (t*16 + rlane), elems j<8 -> K = kb + j, j>=8 -> K = kb + 16 + (j-8)
      v16h b;
      const int col = t * 16 + rlane;
      const int kb  = s * 32 + hi * 8;
#pragma unroll
      for (int j = 0; j < 16; ++j)
        b[j] = (_Float16)W[(size_t)(kb + (j & 7) + ((j >> 3) << 4)) * COUT + col];
      acc[t] = __builtin_amdgcn_wmma_f32_16x16x32_f16(
          /*neg_a=*/false, a[s], /*neg_b=*/false, b,
          /*c_mod=*/(short)0, acc[t], /*reuse_a=*/false, /*reuse_b=*/false);
    }
  }

  // C/D layout: VGPR v -> row (v + 8*hi), col = rlane (within tile)
#pragma unroll
  for (int t = 0; t < NT; ++t) {
    const int col = t * 16 + rlane;
#pragma unroll
    for (int v = 0; v < 8; ++v) {
      const int m = tile * 16 + v + hi * 8;
      if (m < n) Y[(size_t)m * COUT + col] = acc[t][v];
    }
  }
}

// ---------------------------------------------------------------------------
// out[i][c] = dinv[i]^2 * xw[i][c] + bias[c]   (self-loop term + bias init)
// ---------------------------------------------------------------------------
template <int C>
__global__ void k_selfbias(const float* __restrict__ dinv,
                           const float* __restrict__ xw,
                           const float* __restrict__ bias,
                           float* __restrict__ out, int n) {
  int t = blockIdx.x * blockDim.x + threadIdx.x;
  if (t >= n * C) return;
  int i = t / C;
  int c = t % C;
  float di = dinv[i];
  out[t] = di * di * xw[t] + bias[c];
}

// ---------------------------------------------------------------------------
// Edge scatter: out[dst] += norm[e] * xw[src], 4 channels per thread
// ---------------------------------------------------------------------------
template <int C>
__global__ void k_scatter(const long long* __restrict__ src,
                          const long long* __restrict__ dst,
                          const float* __restrict__ norm,
                          const float* __restrict__ xw,
                          float* __restrict__ out, int nE) {
  constexpr int CH4 = C / 4;
  int t = blockIdx.x * blockDim.x + threadIdx.x;
  if (t >= nE * CH4) return;
  int e  = t / CH4;
  int cb = (t % CH4) * 4;
  long long s = src[e], d = dst[e];
  float nv = norm[e];
  float4 v = *(const float4*)(xw + (size_t)s * C + cb);
  float* o = out + (size_t)d * C + cb;
  atomicAdd(o + 0, nv * v.x);
  atomicAdd(o + 1, nv * v.y);
  atomicAdd(o + 2, nv * v.z);
  atomicAdd(o + 3, nv * v.w);
}

__global__ void k_relu(float* __restrict__ h, int n) {
  int t = blockIdx.x * blockDim.x + threadIdx.x;
  if (t < n) h[t] = fmaxf(h[t], 0.f);
}

// ---------------------------------------------------------------------------
static inline unsigned cdiv(long long a, long long b) {
  return (unsigned)((a + b - 1) / b);
}

extern "C" void kernel_launch(void* const* d_in, const int* in_sizes, int n_in,
                              void* d_out, int out_size, void* d_ws, size_t ws_size,
                              hipStream_t stream) {
  const int N = in_sizes[0] / 64;   // x is [N, 64]
  const int E = in_sizes[2];        // edge_weight is [E]

  const float*     x   = (const float*)d_in[0];
  const long long* ei  = (const long long*)d_in[1];  // int64 [2, E]
  const long long* src = ei;
  const long long* dst = ei + E;
  const float*     ew  = (const float*)d_in[2];
  const float*     W1  = (const float*)d_in[3];
  const float*     b1  = (const float*)d_in[4];
  const float*     W2  = (const float*)d_in[5];
  const float*     b2  = (const float*)d_in[6];
  float*           out = (float*)d_out;

  // Workspace carve-up (all f32): deg/dinv[N] | norm[E] | xw[N*64] | h[N*64]
  float* deg = (float*)d_ws;
  size_t off = ((size_t)N + 255) & ~(size_t)255;
  float* nrm = deg + off;
  off += ((size_t)E + 255) & ~(size_t)255;
  float* xw = deg + off;
  off += (size_t)N * 64;
  float* h = deg + off;

  const int nTiles = (N + 15) / 16;

  // Shared degree / normalization precompute
  k_init_deg<<<cdiv(N, 256), 256, 0, stream>>>(deg, N);
  k_deg_scatter<<<cdiv(E, 256), 256, 0, stream>>>(dst, ew, deg, E);
  k_dinv<<<cdiv(N, 256), 256, 0, stream>>>(deg, N);
  k_norm<<<cdiv(E, 256), 256, 0, stream>>>(src, dst, ew, deg, nrm, E);

  // Layer 1: xw = x @ W1 ; h = D^-1/2 (A+I) D^-1/2 xw + b1 ; relu
  k_gemm<64><<<cdiv(nTiles, 4), 128, 0, stream>>>(x, W1, xw, N, nTiles);
  k_selfbias<64><<<cdiv((long long)N * 64, 256), 256, 0, stream>>>(deg, xw, b1, h, N);
  k_scatter<64><<<cdiv((long long)E * 16, 256), 256, 0, stream>>>(src, dst, nrm, xw, h, E);
  k_relu<<<cdiv((long long)N * 64, 256), 256, 0, stream>>>(h, N * 64);

  // Layer 2: xw = h @ W2 ; out = D^-1/2 (A+I) D^-1/2 xw + b2
  k_gemm<32><<<cdiv(nTiles, 4), 128, 0, stream>>>(h, W2, xw, N, nTiles);
  k_selfbias<32><<<cdiv((long long)N * 32, 256), 256, 0, stream>>>(deg, xw, b2, out, N);
  k_scatter<32><<<cdiv((long long)E * 8, 256), 256, 0, stream>>>(src, dst, nrm, xw, out, E);
}
